// IndRNNNet_69939247448811
// MI455X (gfx1250) — compile-verified, matrix operand order checked by reading, not run on hardware
//
#include <hip/hip_runtime.h>

// ---------- types ----------
typedef __attribute__((ext_vector_type(16))) __bf16        v16bf;
typedef __attribute__((ext_vector_type(8)))  float         v8f;
typedef __attribute__((ext_vector_type(4)))  unsigned int  u32x4;
typedef __attribute__((ext_vector_type(2)))  float         f32x2;

static __device__ __forceinline__ unsigned short f2bf(float f) {
  unsigned int u = __float_as_uint(f);
  unsigned int r = (u >> 16) & 1u;       // round-to-nearest-even
  u += 0x7fffu + r;
  return (unsigned short)(u >> 16);
}

// ---------- f32 -> bf16 conversion (2 elements / thread) ----------
__global__ void cvt_f32_bf16_kernel(const float* __restrict__ src,
                                    unsigned short* __restrict__ dst,
                                    long long n2) {
  long long i = (long long)blockIdx.x * blockDim.x + threadIdx.x;
  if (i >= n2) return;
  f32x2 v = ((const f32x2*)src)[i];
  unsigned int p = (unsigned int)f2bf(v.x) | ((unsigned int)f2bf(v.y) << 16);
  ((unsigned int*)dst)[i] = p;
}

// ---------- IndRNN elementwise scan: h_t = relu(p_t + u*h_{t-1}), emit bf16 ----------
// One thread per (batch, hidden) chain; software-pipelined prefetch of 16
// timesteps keeps up to 32 global loads in flight per wave (we run ~1 wave
// per SIMD here, so per-wave MLP is the whole ballgame).
#define SG 16   // scan group size (T % (2*SG) == 0)

__global__ void indrnn_scan_kernel(const float* __restrict__ proj,
                                   const float* __restrict__ u,
                                   unsigned short* __restrict__ hout,
                                   int T, int H) {
  int idx = blockIdx.x * blockDim.x + threadIdx.x;   // b*H + h within chunk
  int b = idx / H;
  int hh = idx - b * H;
  const float uu = u[hh];
  const size_t base = (size_t)b * T * H + hh;
  float h = 0.f;

  float p[2][SG];
  const int ngrp = T / SG;

  // prologue: prefetch group 0
#pragma unroll
  for (int i = 0; i < SG; ++i) p[0][i] = proj[base + (size_t)i * H];

  for (int g = 0; g < ngrp; ++g) {
    const int cur = g & 1;
    // prefetch next group while current group's recurrence runs
    if (g + 1 < ngrp) {
      const size_t nb = base + (size_t)(g + 1) * SG * H;
#pragma unroll
      for (int i = 0; i < SG; ++i) p[cur ^ 1][i] = proj[nb + (size_t)i * H];
    }
    const size_t cb = base + (size_t)g * SG * H;
#pragma unroll
    for (int i = 0; i < SG; ++i) {
      h = fmaf(uu, h, p[cur][i]);
      h = h > 0.f ? h : 0.f;
      hout[cb + (size_t)i * H] = f2bf(h);
    }
  }
}

// ---------- WMMA GEMM: C[M,N] (f32) = A[M,K](bf16) * W[N,K](bf16)^T + bias1 + bias2 ----------
#define KT     32
#define LDSROW 40   // padded row (ushorts) to spread LDS banks

__global__ __launch_bounds__(256)
void wmma_gemm_bf16_kernel(const unsigned short* __restrict__ A,
                           const unsigned short* __restrict__ W,
                           float* __restrict__ C,
                           const float* __restrict__ bias1,
                           const float* __restrict__ bias2,
                           int M, int N, int K) {
  __shared__ unsigned short lds[2][2][128][LDSROW];   // [buf][A/B][row][k] = 40 KB
  const int tid  = threadIdx.x;
  const int lane = tid & 31;
  const int wave = tid >> 5;
  const int wm   = wave >> 1;          // 0..3 : M sub-tile (32 rows)
  const int wn   = wave & 1;           // 0..1 : N sub-tile (64 cols)
  const int lm   = lane & 15;
  const int hl   = lane >> 4;          // lane half, selects K group per WMMA layout
  const int m0   = blockIdx.y * 128;
  const int n0   = blockIdx.x * 128;

  // global->LDS staging: 256 threads, 16B units; each thread moves 2 rows per matrix
  const int r0 = tid >> 2;
  const int c0 = (tid & 3) * 8;        // ushort offset within 32-wide K slab

  u32x4 ra0, ra1, rb0, rb1;
  auto gload = [&](int k0) {
    ra0 = *(const u32x4*)&A[(size_t)(m0 + r0)      * K + k0 + c0];
    ra1 = *(const u32x4*)&A[(size_t)(m0 + r0 + 64) * K + k0 + c0];
    rb0 = *(const u32x4*)&W[(size_t)(n0 + r0)      * K + k0 + c0];
    rb1 = *(const u32x4*)&W[(size_t)(n0 + r0 + 64) * K + k0 + c0];
  };
  auto stlds = [&](int buf) {
    *(u32x4*)&lds[buf][0][r0][c0]      = ra0;
    *(u32x4*)&lds[buf][0][r0 + 64][c0] = ra1;
    *(u32x4*)&lds[buf][1][r0][c0]      = rb0;
    *(u32x4*)&lds[buf][1][r0 + 64][c0] = rb1;
  };

  v8f acc[2][4] = {};
  union Frag { v16bf v; u32x4 q[2]; };

  gload(0);
  stlds(0);
  __syncthreads();

  const int nsteps = K / KT;
  for (int s = 0; s < nsteps; ++s) {
    const int buf = s & 1;
    if (s + 1 < nsteps) gload((s + 1) * KT);

    // Fragment loads per CDNA5 16-bit A/B layout:
    //   VGPRs 0..3 hold K pairs {2v + 8*hl}  -> contiguous 16B at ushort ofs 8*hl
    //   VGPRs 4..7 hold K pairs {16 + ...}   -> contiguous 16B at ushort ofs 16 + 8*hl
    Frag af[2], bfr[4];
#pragma unroll
    for (int tm = 0; tm < 2; ++tm) {
      int r = wm * 32 + tm * 16 + lm;
      af[tm].q[0] = *(const u32x4*)&lds[buf][0][r][hl * 8];
      af[tm].q[1] = *(const u32x4*)&lds[buf][0][r][16 + hl * 8];
    }
#pragma unroll
    for (int tn = 0; tn < 4; ++tn) {
      int r = wn * 64 + tn * 16 + lm;
      bfr[tn].q[0] = *(const u32x4*)&lds[buf][1][r][hl * 8];
      bfr[tn].q[1] = *(const u32x4*)&lds[buf][1][r][16 + hl * 8];
    }
#pragma unroll
    for (int tm = 0; tm < 2; ++tm)
#pragma unroll
      for (int tn = 0; tn < 4; ++tn)
        acc[tm][tn] = __builtin_amdgcn_wmma_f32_16x16x32_bf16(
            false, af[tm].v, false, bfr[tn].v, (short)0, acc[tm][tn],
            false, false);

    if (s + 1 < nsteps) stlds(buf ^ 1);
    __syncthreads();
  }

  // Epilogue: C/D layout -> lane (hl,lm): N = lm, M = v + 8*hl
#pragma unroll
  for (int tn = 0; tn < 4; ++tn) {
    const int gc = n0 + wn * 64 + tn * 16 + lm;
    float bias = 0.f;
    if (bias1) bias += bias1[gc];
    if (bias2) bias += bias2[gc];
#pragma unroll
    for (int tm = 0; tm < 2; ++tm) {
      const int gm = m0 + wm * 32 + tm * 16 + hl * 8;
#pragma unroll
      for (int v = 0; v < 8; ++v)
        C[(size_t)(gm + v) * N + gc] = acc[tm][tn][v] + bias;
    }
  }
}

// ---------- driver ----------
extern "C" void kernel_launch(void* const* d_in, const int* in_sizes, int n_in,
                              void* d_out, int out_size, void* d_ws, size_t ws_size,
                              hipStream_t stream) {
  (void)in_sizes; (void)n_in; (void)out_size;
  const int B = 64, T = 2048, Din = 128, H = 512, Dout = 128;
  const float* x   = (const float*)d_in[0];
  const float* W0  = (const float*)d_in[1];
  const float* bl0 = (const float*)d_in[2];
  const float* u0  = (const float*)d_in[3];
  const float* bb0 = (const float*)d_in[4];
  const float* W1  = (const float*)d_in[5];
  const float* bl1 = (const float*)d_in[6];
  const float* u1  = (const float*)d_in[7];
  const float* bb1 = (const float*)d_in[8];
  const float* fcW = (const float*)d_in[9];
  const float* fcb = (const float*)d_in[10];
  float* out = (float*)d_out;

  char* ws = (char*)d_ws;
  unsigned short* W0b = (unsigned short*)ws;
  unsigned short* W1b = W0b + (size_t)H * Din;
  unsigned short* fWb = W1b + (size_t)H * H;
  size_t woff = ((size_t)H * Din + (size_t)H * H + (size_t)Dout * H) * 2;
  woff = (woff + 255) & ~(size_t)255;

  auto cvt = [&](const float* s, unsigned short* d, long long n) {
    long long n2 = n / 2;
    cvt_f32_bf16_kernel<<<(unsigned)((n2 + 255) / 256), 256, 0, stream>>>(s, d, n2);
  };

  // convert weights once (already [N, K] row-major, matching B-operand addressing)
  cvt(W0,  W0b, (long long)H * Din);
  cvt(W1,  W1b, (long long)H * H);
  cvt(fcW, fWb, (long long)Dout * H);

  // chunk batch to fit workspace: per batch needs T*H*4 (proj f32) + T*H*2 (h bf16)
  size_t perBatch = (size_t)T * H * 6;
  size_t avail = ws_size > woff ? ws_size - woff : 0;
  int CB = (int)(avail / perBatch);
  if (CB < 1) CB = 1;
  if (CB > B) CB = B;

  float*          regA = (float*)(ws + woff);                                   // proj f32
  unsigned short* regB = (unsigned short*)(ws + woff + (size_t)CB * T * H * 4); // bf16 acts

  for (int b0 = 0; b0 < B; b0 += CB) {
    const int cb = (B - b0 < CB) ? (B - b0) : CB;
    const long long Mrows = (long long)cb * T;

    // 1) x chunk -> bf16
    cvt(x + (size_t)b0 * T * Din, regB, Mrows * Din);

    // 2) layer0 GEMM: proj0 = xbf @ W0^T + bl0 + bb0
    dim3 g0(H / 128, (unsigned)(Mrows / 128));
    wmma_gemm_bf16_kernel<<<g0, 256, 0, stream>>>(regB, W0b, regA, bl0, bb0,
                                                  (int)Mrows, H, Din);
    // 3) scan0 -> h1 (bf16)
    indrnn_scan_kernel<<<(unsigned)(cb * H / 256), 256, 0, stream>>>(regA, u0, regB, T, H);

    // 4) layer1 GEMM: proj1 = h1 @ W1^T + bl1 + bb1
    wmma_gemm_bf16_kernel<<<g0, 256, 0, stream>>>(regB, W1b, regA, bl1, bb1,
                                                  (int)Mrows, H, H);
    // 5) scan1 -> h2 (bf16)
    indrnn_scan_kernel<<<(unsigned)(cb * H / 256), 256, 0, stream>>>(regA, u1, regB, T, H);

    // 6) head GEMM: out = h2 @ fcW^T + fc_b
    dim3 g2(Dout / 128, (unsigned)(Mrows / 128));
    wmma_gemm_bf16_kernel<<<g2, 256, 0, stream>>>(regB, fWb,
                                                  out + (size_t)b0 * T * Dout,
                                                  fcb, nullptr, (int)Mrows, Dout, H);
  }
}
